// MultiHeadAttentionLoRA_60129542144249
// MI455X (gfx1250) — matrix operand order repaired and checked by reading, not compiled, not run on hardware
//
#include <hip/hip_runtime.h>

typedef __attribute__((ext_vector_type(16))) _Float16 v16h;
typedef __attribute__((ext_vector_type(8)))  _Float16 v8h;
typedef __attribute__((ext_vector_type(8)))  float    v8f;
typedef _Float16 half_t;

#define B_    64
#define SEQ   197
#define D_    768
#define H_    12
#define HD    64
#define M_TOT (B_*SEQ)   /* 12608, divisible by 64 */
#define NPITCH 224       /* padded key dim for transposed V */
#define APITCH 40        /* LDS row pitch in halfs (80B) to spread banks */
#define LORA_SCALE 2.0f  /* alpha/r = 16/8 */

#define USE_ASYNC_LDS 1  /* global_load_async_to_lds_b128 path (gfx1250) */

// ---------------------------------------------------------------------------
// WMMA fragment helpers (CDNA5 16x16x32 f16 layouts, wave32)
// A 16x32: lane&15 = M; halves 0..7 -> K = hi*8 .. hi*8+7 ; halves 8..15 -> +16
// B 32x16: lane&15 = N; halves 0..7 -> K = hi*8 .. hi*8+7 ; halves 8..15 -> +16
// C 16x16: elem j -> M = j + hi*8, N = lane&15
// ---------------------------------------------------------------------------
__device__ __forceinline__ v16h fragA16(const half_t* rowptr, int laneHi) {
  v8h lo = *(const v8h*)(rowptr + laneHi * 8);
  v8h hi = *(const v8h*)(rowptr + laneHi * 8 + 16);
  return __builtin_shufflevector(lo, hi, 0,1,2,3,4,5,6,7,8,9,10,11,12,13,14,15);
}
__device__ __forceinline__ v16h fragB16(const half_t* rowptr, int laneHi) {
  v8h lo = *(const v8h*)(rowptr + laneHi * 8);
  v8h hi = *(const v8h*)(rowptr + laneHi * 8 + 16);
  return __builtin_shufflevector(lo, hi, 0,1,2,3,4,5,6,7,8,9,10,11,12,13,14,15);
}
__device__ __forceinline__ v8f wmma_f16(v16h a, v16h b, v8f c) {
  return __builtin_amdgcn_wmma_f32_16x16x32_f16(false, a, false, b, (short)0, c, false, false);
}

#if USE_ASYNC_LDS
// Copy 16B/lane global -> LDS without VGPR round trip (ASYNCcnt-tracked).
// VDST operand = per-lane LDS byte offset = low 32 bits of generic shared ptr.
__device__ __forceinline__ void async_cp16(const half_t* g, half_t* lds) {
  unsigned l = (unsigned)(uintptr_t)lds;
  unsigned long long ga = (unsigned long long)(uintptr_t)g;
  asm volatile("global_load_async_to_lds_b128 %0, %1, off"
               :: "v"(l), "v"(ga) : "memory");
}
__device__ __forceinline__ void wait_async0() {
  asm volatile("s_wait_asynccnt 0x0" ::: "memory");
}
#endif

// ---------------------------------------------------------------------------
// Prep kernels
// ---------------------------------------------------------------------------
__global__ void cvt_f32_to_f16(const float* __restrict__ src,
                               half_t* __restrict__ dst, int n) {
  int i = blockIdx.x * 256 + threadIdx.x;
  if (i < n) dst[i] = (half_t)src[i];
}

__global__ void fill_zero_f16(half_t* __restrict__ dst, int n) {
  int i = blockIdx.x * 256 + threadIdx.x;
  if (i < n) dst[i] = (half_t)0.0f;
}

// W_eff[o,d] = w[o,d] + scale * sum_r lb[o,r]*la[r,d]   (exact fp32 fold, then f16)
__global__ void build_weff(const float* __restrict__ w, const float* __restrict__ la,
                           const float* __restrict__ lb, half_t* __restrict__ weff,
                           int O, int D) {
  int i = blockIdx.x * 256 + threadIdx.x;
  if (i >= O * D) return;
  int o = i / D, d = i - o * D;
  float s = 0.f;
#pragma unroll
  for (int r = 0; r < 8; ++r) s += lb[o * 8 + r] * la[r * D + d];
  weff[i] = (half_t)(w[i] + LORA_SCALE * s);
}

// ---------------------------------------------------------------------------
// Tiled f16 WMMA GEMM: C[M,N] = A[M,K] @ W[N,K]^T + bias
// block = 128 threads (4 waves); tile 64x64; K-step 32
// mode 0: scatter f16 into qkbuf (q,k) / vbuf (v transposed, key-padded)
// mode 1: f32 straight to out
// ---------------------------------------------------------------------------
__global__ __launch_bounds__(128)
void gemm_f16_wmma(const half_t* __restrict__ A, const half_t* __restrict__ W,
                   const float* __restrict__ bias, int K, int N, int mode,
                   half_t* __restrict__ qkbuf, half_t* __restrict__ vbuf,
                   float* __restrict__ outf) {
  __shared__ __align__(16) half_t As[64 * APITCH];
  __shared__ __align__(16) half_t Ws[64 * APITCH];

  const int tid    = threadIdx.x;
  const int lane   = tid & 31;
  const int wave   = tid >> 5;
  const int laneHi = lane >> 4;
  const int laneLo = lane & 15;
  const int n0 = blockIdx.x * 64;
  const int m0 = blockIdx.y * 64;

  v8f acc[4];
#pragma unroll
  for (int i = 0; i < 4; ++i)
#pragma unroll
    for (int j = 0; j < 8; ++j) acc[i][j] = 0.0f;

  const int lr = tid >> 1;          // 0..63 tile row loaded by this thread
  const int lc = (tid & 1) * 16;    // half-column 0 or 16

  for (int kk = 0; kk < K; kk += 32) {
    const half_t* ga = A + (size_t)(m0 + lr) * K + kk + lc;
    const half_t* gw = W + (size_t)(n0 + lr) * K + kk + lc;
    half_t* la = As + lr * APITCH + lc;
    half_t* lw = Ws + lr * APITCH + lc;
#if USE_ASYNC_LDS
    __syncthreads();                       // prior tile fully consumed
    async_cp16(ga,     la);
    async_cp16(ga + 8, la + 8);
    async_cp16(gw,     lw);
    async_cp16(gw + 8, lw + 8);
    __builtin_prefetch(ga + 32, 0, 3);     // global_prefetch_b8: next K tile
    __builtin_prefetch(gw + 32, 0, 3);
    wait_async0();                         // own copies landed in LDS
    __syncthreads();                       // everyone's copies visible
#else
    v8h a0 = *(const v8h*)ga;
    v8h a1 = *(const v8h*)(ga + 8);
    v8h w0 = *(const v8h*)gw;
    v8h w1 = *(const v8h*)(gw + 8);
    __builtin_prefetch(ga + 32, 0, 3);
    __builtin_prefetch(gw + 32, 0, 3);
    __syncthreads();
    *(v8h*)la = a0; *(v8h*)(la + 8) = a1;
    *(v8h*)lw = w0; *(v8h*)(lw + 8) = w1;
    __syncthreads();
#endif

    // Batch all fragment loads, then issue the 4 WMMAs back-to-back.
    v16h af = fragA16(As + (wave * 16 + laneLo) * APITCH, laneHi);
    v16h bf[4];
#pragma unroll
    for (int nt = 0; nt < 4; ++nt)
      bf[nt] = fragB16(Ws + (nt * 16 + laneLo) * APITCH, laneHi);
#pragma unroll
    for (int nt = 0; nt < 4; ++nt)
      acc[nt] = wmma_f16(af, bf[nt], acc[nt]);
  }

  const int mbase = m0 + wave * 16 + laneHi * 8;
#pragma unroll
  for (int nt = 0; nt < 4; ++nt) {
    int o = n0 + nt * 16 + laneLo;
    float bv = bias[o];
#pragma unroll
    for (int j = 0; j < 8; ++j) {
      int m = mbase + j;
      float v = acc[nt][j] + bv;
      if (mode == 1) {
        outf[(size_t)m * N + o] = v;
      } else {
        int b = m / SEQ, nn = m - b * SEQ;
        int which = o / D_;
        int rem = o - which * D_;
        int h = rem >> 6, d = rem & 63;
        if (which < 2) { // Q,K: [which][b][h][n][64]
          qkbuf[((((size_t)which * B_ + b) * H_ + h) * SEQ + nn) * HD + d] = (half_t)v;
        } else {         // V transposed: [b][h][d][NPITCH]
          vbuf[(((size_t)b * H_ + h) * HD + d) * NPITCH + nn] = (half_t)v;
        }
      }
    }
  }
}

// ---------------------------------------------------------------------------
// Flash-style attention: one wave per (qtile, h, b). 16 query rows.
// 7 key chunks of 32; online softmax with 16-lane shfl_xor row reductions.
// ---------------------------------------------------------------------------
__global__ __launch_bounds__(32)
void attn_wmma(const half_t* __restrict__ qkbuf, const half_t* __restrict__ vbuf,
               half_t* __restrict__ attnbuf) {
  __shared__ __align__(16) half_t Pb[16 * APITCH];

  const int lane   = threadIdx.x & 31;
  const int laneHi = lane >> 4;
  const int laneLo = lane & 15;
  const int qt = blockIdx.x, h = blockIdx.y, b = blockIdx.z;

  const half_t* qp = qkbuf + (((size_t)(0 * B_ + b) * H_ + h) * SEQ) * HD;
  const half_t* kp = qkbuf + (((size_t)(1 * B_ + b) * H_ + h) * SEQ) * HD;
  const half_t* vp = vbuf  + (((size_t)b * H_ + h) * HD) * NPITCH;

  int qr = qt * 16 + laneLo;
  int qrc = qr < SEQ - 1 ? qr : SEQ - 1;
  v16h qf0 = fragA16(qp + (size_t)qrc * HD + 0,  laneHi);
  v16h qf1 = fragA16(qp + (size_t)qrc * HD + 32, laneHi);

  float mrow[8], lrow[8];
  v8f o[4];
#pragma unroll
  for (int j = 0; j < 8; ++j) { mrow[j] = -1e30f; lrow[j] = 0.0f; }
#pragma unroll
  for (int nt = 0; nt < 4; ++nt)
#pragma unroll
    for (int j = 0; j < 8; ++j) o[nt][j] = 0.0f;

  for (int it = 0; it < 7; ++it) {
    const int c0 = it * 32;
    v8f s0, s1;
#pragma unroll
    for (int j = 0; j < 8; ++j) { s0[j] = 0.0f; s1[j] = 0.0f; }

    int k0 = c0 + laneLo;       int k0c = k0 < SEQ - 1 ? k0 : SEQ - 1;
    int k1 = c0 + 16 + laneLo;  int k1c = k1 < SEQ - 1 ? k1 : SEQ - 1;
    v16h kf[4];
    kf[0] = fragB16(kp + (size_t)k0c * HD + 0,  laneHi);
    kf[1] = fragB16(kp + (size_t)k0c * HD + 32, laneHi);
    kf[2] = fragB16(kp + (size_t)k1c * HD + 0,  laneHi);
    kf[3] = fragB16(kp + (size_t)k1c * HD + 32, laneHi);
    s0 = wmma_f16(qf0, kf[0], s0);
    s0 = wmma_f16(qf1, kf[1], s0);
    s1 = wmma_f16(qf0, kf[2], s1);
    s1 = wmma_f16(qf1, kf[3], s1);

    const bool v0 = k0 < SEQ, v1 = k1 < SEQ;
    float sc0[8], sc1[8], tm[8];
#pragma unroll
    for (int j = 0; j < 8; ++j) {
      sc0[j] = v0 ? s0[j] * 0.125f : -1e30f;   // Hd^-0.5 = 1/8
      sc1[j] = v1 ? s1[j] * 0.125f : -1e30f;
    }
#pragma unroll
    for (int j = 0; j < 8; ++j) {
      float t = fmaxf(sc0[j], sc1[j]);
#pragma unroll
      for (int x = 1; x < 16; x <<= 1) t = fmaxf(t, __shfl_xor(t, x, 32));
      tm[j] = t;
    }
    float p0[8], p1[8], scl[8];
#pragma unroll
    for (int j = 0; j < 8; ++j) {
      float mn = fmaxf(mrow[j], tm[j]);
      scl[j] = __expf(mrow[j] - mn);
      mrow[j] = mn;
      p0[j] = __expf(sc0[j] - mn);
      p1[j] = __expf(sc1[j] - mn);
    }
#pragma unroll
    for (int j = 0; j < 8; ++j) {
      float t = p0[j] + p1[j];
#pragma unroll
      for (int x = 1; x < 16; x <<= 1) t += __shfl_xor(t, x, 32);
      lrow[j] = lrow[j] * scl[j] + t;
    }
#pragma unroll
    for (int nt = 0; nt < 4; ++nt)
#pragma unroll
      for (int j = 0; j < 8; ++j) o[nt][j] *= scl[j];

    __syncthreads();
#pragma unroll
    for (int j = 0; j < 8; ++j) {
      Pb[(j + laneHi * 8) * APITCH + laneLo]      = (half_t)p0[j];
      Pb[(j + laneHi * 8) * APITCH + 16 + laneLo] = (half_t)p1[j];
    }
    __syncthreads();

    v16h pf = fragA16(Pb + laneLo * APITCH, laneHi);
    v16h vf[4];
#pragma unroll
    for (int nt = 0; nt < 4; ++nt)
      vf[nt] = fragB16(vp + (size_t)(nt * 16 + laneLo) * NPITCH + c0, laneHi);
#pragma unroll
    for (int nt = 0; nt < 4; ++nt)
      o[nt] = wmma_f16(pf, vf[nt], o[nt]);
  }

  const int rowbase = qt * 16 + laneHi * 8;
#pragma unroll
  for (int nt = 0; nt < 4; ++nt)
#pragma unroll
    for (int j = 0; j < 8; ++j) {
      int row = rowbase + j;
      if (row < SEQ) {
        float val = o[nt][j] / lrow[j];
        attnbuf[((size_t)b * SEQ + row) * D_ + h * HD + nt * 16 + laneLo] = (half_t)val;
      }
    }
}

// ---------------------------------------------------------------------------
extern "C" void kernel_launch(void* const* d_in, const int* in_sizes, int n_in,
                              void* d_out, int out_size, void* d_ws, size_t ws_size,
                              hipStream_t stream) {
  (void)in_sizes; (void)n_in; (void)out_size; (void)ws_size;
  const float* x   = (const float*)d_in[0];
  const float* qw  = (const float*)d_in[1];
  const float* qb  = (const float*)d_in[2];
  const float* qla = (const float*)d_in[3];
  const float* qlb = (const float*)d_in[4];
  const float* ow  = (const float*)d_in[5];
  const float* ob  = (const float*)d_in[6];
  const float* ola = (const float*)d_in[7];
  const float* olb = (const float*)d_in[8];
  float* out = (float*)d_out;

  half_t* p  = (half_t*)d_ws;
  half_t* xh = p;  p += (size_t)M_TOT * D_;        // x in f16; later reused as attnbuf
  half_t* w1 = p;  p += (size_t)3 * D_ * D_;       // qkv W_eff
  half_t* w2 = p;  p += (size_t)D_ * D_;           // out W_eff
  half_t* qk = p;  p += (size_t)2 * B_ * H_ * SEQ * HD;  // Q,K
  half_t* vb = p;  p += (size_t)B_ * H_ * HD * NPITCH;   // V transposed, padded
  half_t* attnbuf = xh;                            // alias: xh dead after GEMM1

  int n;
  n = M_TOT * D_;
  cvt_f32_to_f16<<<(n + 255) / 256, 256, 0, stream>>>(x, xh, n);
  n = 3 * D_ * D_;
  build_weff<<<(n + 255) / 256, 256, 0, stream>>>(qw, qla, qlb, w1, 3 * D_, D_);
  n = D_ * D_;
  build_weff<<<(n + 255) / 256, 256, 0, stream>>>(ow, ola, olb, w2, D_, D_);
  n = B_ * H_ * HD * NPITCH;
  fill_zero_f16<<<(n + 255) / 256, 256, 0, stream>>>(vb, n);

  gemm_f16_wmma<<<dim3(3 * D_ / 64, M_TOT / 64), 128, 0, stream>>>(
      xh, w1, qb, D_, 3 * D_, 0, qk, vb, nullptr);

  attn_wmma<<<dim3((SEQ + 15) / 16, H_, B_), 32, 0, stream>>>(qk, vb, attnbuf);

  gemm_f16_wmma<<<dim3(D_ / 64, M_TOT / 64), 128, 0, stream>>>(
      attnbuf, w2, ob, D_, D_, 1, nullptr, nullptr, out);
}
